// MultiHeadSelfAttention_65051574665255
// MI455X (gfx1250) — compile-verified, matrix operand order checked by reading, not run on hardware
//
#include <hip/hip_runtime.h>

// ---------------------------------------------------------------------------
// MI455X (gfx1250) multi-head self-attention, f16 WMMA path.
// out  = (ctx @ Wo^T + bo)                  [B,S,D]  f32  -> d_out[0 : B*S*D]
// ctx  = softmax(Q K^T / sqrt(HD)) V        [B,H,S,HD] f32 -> d_out[B*S*D : ]
// Matmuls on v_wmma_f32_16x16x32_f16; LDS staging via CDNA5 async
// global->LDS DMA (global_load_async_to_lds_b128, ASYNCcnt) + double buffer.
// Softmax row sums via WMMA against a ones-matrix; row max via batched
// ds_swizzle XOR reductions. K/V fragments preloaded in bulk so DS waits
// stage instead of serializing against each WMMA.
// ---------------------------------------------------------------------------

#define DEV_INLINE __device__ __forceinline__

typedef __attribute__((ext_vector_type(16))) _Float16 v16h;
typedef __attribute__((ext_vector_type(8)))  _Float16 v8h;
typedef __attribute__((ext_vector_type(4)))  _Float16 v4h;
typedef __attribute__((ext_vector_type(8)))  float    v8f;
typedef __attribute__((ext_vector_type(4)))  float    v4f;

constexpr int Bk = 4, Sk = 2048, Dk = 1024, Hk = 16, HDk = 64;
constexpr int Mk = Bk * Sk;  // 8192 rows

union H16 { v16h v; v8h h[2]; };

DEV_INLINE v8f wmma16(v16h a, v16h b, v8f c) {
  // (neg_a, A, neg_b, B, c_mod, C, reuse_a, reuse_b)
  return __builtin_amdgcn_wmma_f32_16x16x32_f16(false, a, false, b, (short)0, c,
                                                false, false);
}

// Fragment chunk loader: caller passes address of (row, k0 + half*8);
// the two 16B chunks at +0 and +16 match the 16-bit A/B VGPR layout
// (lanes 0-15: K = k0..k0+7 & k0+16..k0+23; lanes 16-31: +8).
DEV_INLINE v16h frag_ld(const _Float16* p0) {
  H16 u;
  u.h[0] = *(const v8h*)(p0);
  u.h[1] = *(const v8h*)(p0 + 16);
  return u.v;
}

// CDNA5 async global->LDS DMA, 16B per lane. LDS byte offset in vdst VGPR,
// 64-bit global address in VGPR pair. Tracked by ASYNCcnt.
DEV_INLINE void async_ld_b128(unsigned lds_off, const void* gptr) {
  asm volatile("global_load_async_to_lds_b128 %0, %1, off"
               :: "v"(lds_off), "v"((unsigned long long)(size_t)gptr)
               : "memory");
}

DEV_INLINE void wait_async0() {
#if __has_builtin(__builtin_amdgcn_s_wait_asynccnt)
  __builtin_amdgcn_s_wait_asynccnt(0);
#else
  asm volatile("s_wait_asynccnt 0x0" ::: "memory");
#endif
}

// Generic shared pointer -> LDS byte offset (low 32 bits per LDS aperture).
DEV_INLINE unsigned lds_addr(const void* p) { return (unsigned)(size_t)p; }

// Batched XOR-swizzle max over 8 independent rows: 8 DS ops in flight per
// step instead of a serial bpermute/wait chain per row.
template <int MASK>
DEV_INLINE void xor_max8(float* pm) {
  float t[8];
#pragma unroll
  for (int r = 0; r < 8; ++r)
    t[r] = __int_as_float(__builtin_amdgcn_ds_swizzle(
        __float_as_int(pm[r]), (MASK << 10) | 0x1F));
#pragma unroll
  for (int r = 0; r < 8; ++r) pm[r] = fmaxf(pm[r], t[r]);
}

// ---------------------------------------------------------------------------
// Weight f32 -> f16 conversion (weights reused by 64 M-tiles; convert once).
// dst = [wq16 | wk16 | wv16 | wo16] contiguous, each D*D.
// ---------------------------------------------------------------------------
__global__ __launch_bounds__(256) void cvt_w_kernel(
    const float* __restrict__ w0, const float* __restrict__ w1,
    const float* __restrict__ w2, const float* __restrict__ w3,
    _Float16* __restrict__ dst) {
  const size_t i4 = ((size_t)blockIdx.x * 256 + threadIdx.x) * 4;
  const size_t which = i4 >> 20;  // D*D == 1<<20
  const size_t off = i4 & ((1u << 20) - 1);
  const float* src = (which == 0) ? w0 : (which == 1) ? w1
                   : (which == 2) ? w2 : w3;
  v4f f = *(const v4f*)(src + off);
  v4h hv;
#pragma unroll
  for (int i = 0; i < 4; ++i) hv[i] = (_Float16)f[i];
  *(v4h*)(dst + i4) = hv;
}

// ---------------------------------------------------------------------------
// Tiled GEMM: Y[M,N] = X[M,K] @ W[N,K]^T + bias[N].
// B-matrix column n == W row n -> contiguous 16B fragment chunks.
// 256 threads = 8 waves; block tile 128x128; K-step 32; double-buffered LDS.
// B tile always async-DMA'd; A tile async when f16, software-pipelined
// load+cvt+ds_store when f32 (QKV projections).
// MODE 0: f16 out, head layout [B,H,S,HD]; MODE 1: f16 out [B,H,HD,S];
// MODE 2: f32 out [M,N] -> d_out.
// ---------------------------------------------------------------------------
template <int AF16, int MODE>
__global__ __launch_bounds__(256) void gemm_xwT_kernel(
    const void* __restrict__ Asrc, const _Float16* __restrict__ W16,
    const float* __restrict__ bias, _Float16* __restrict__ out16,
    float* __restrict__ out32) {
  constexpr int TM = 128, TN = 128, TK = 32, NK = Dk / TK;
  constexpr int LDA = 40, LDB = 40;  // +8 f16 pad: 80B rows, b128-conflict-free
  __shared__ _Float16 As[2][TM * LDA];
  __shared__ _Float16 Bs[2][TN * LDB];

  const int tid = threadIdx.x;
  const int wave = tid >> 5, lid = tid & 31;
  const int half = lid >> 4, l16 = lid & 15;
  const int wm = (wave >> 2) * 64;  // 0 / 64
  const int wn = (wave & 3) * 32;   // 0 / 32 / 64 / 96
  const int m0 = blockIdx.y * TM;
  const int n0 = blockIdx.x * TN;

  const int lr = tid >> 1;        // tile row 0..127
  const int lc = (tid & 1) << 4;  // col 0 / 16

  const _Float16* Wrow = W16 + (size_t)(n0 + lr) * Dk + lc;

  auto stageB = [&](int buf, int k0) {
    const unsigned d = lds_addr(&Bs[buf][lr * LDB + lc]);
    async_ld_b128(d, Wrow + k0);
    async_ld_b128(d + 16, Wrow + k0 + 8);
  };
  auto stageA16 = [&](int buf, int k0) {
    const _Float16* ap = (const _Float16*)Asrc + (size_t)(m0 + lr) * Dk + lc;
    const unsigned d = lds_addr(&As[buf][lr * LDA + lc]);
    async_ld_b128(d, ap + k0);
    async_ld_b128(d + 16, ap + k0 + 8);
  };
  auto issueA32 = [&](int k0, v4f* f) {
    const float* ap = (const float*)Asrc + (size_t)(m0 + lr) * Dk + k0 + lc;
#pragma unroll
    for (int i = 0; i < 4; ++i) f[i] = *(const v4f*)(ap + 4 * i);
  };
  auto commitA32 = [&](int buf, const v4f* f) {
    v8h h0, h1;
#pragma unroll
    for (int i = 0; i < 4; ++i) {
      h0[i] = (_Float16)f[0][i]; h0[4 + i] = (_Float16)f[1][i];
      h1[i] = (_Float16)f[2][i]; h1[4 + i] = (_Float16)f[3][i];
    }
    *(v8h*)&As[buf][lr * LDA + lc] = h0;
    *(v8h*)&As[buf][lr * LDA + lc + 8] = h1;
  };

  v8f zero = {};
  v8f acc[4][2];
#pragma unroll
  for (int i = 0; i < 4; ++i)
#pragma unroll
    for (int j = 0; j < 2; ++j) acc[i][j] = zero;

  // -------- prologue: fill buffer 0
  if constexpr (AF16) {
    stageA16(0, 0);
  } else {
    v4f f[4];
    issueA32(0, f);
    commitA32(0, f);
  }
  stageB(0, 0);
  wait_async0();
  __syncthreads();

  // -------- main pipeline: one barrier per K-step; unroll x2 so the
  // double-buffer index constant-folds.
#pragma unroll 2
  for (int k = 0; k < NK; ++k) {
    const int cur = k & 1;
    const bool more = (k + 1 < NK);
    v4f fnext[4];
    if (more) {
      if constexpr (AF16) stageA16(cur ^ 1, (k + 1) * TK);
      else issueA32((k + 1) * TK, fnext);
      stageB(cur ^ 1, (k + 1) * TK);
    }

    const _Float16* Ab = As[cur];
    const _Float16* Bb = Bs[cur];
    v16h aF[4], bF[2];
#pragma unroll
    for (int mf = 0; mf < 4; ++mf)
      aF[mf] = frag_ld(&Ab[(wm + mf * 16 + l16) * LDA + (half << 3)]);
#pragma unroll
    for (int nf = 0; nf < 2; ++nf)
      bF[nf] = frag_ld(&Bb[(wn + nf * 16 + l16) * LDB + (half << 3)]);
#pragma unroll
    for (int mf = 0; mf < 4; ++mf)
#pragma unroll
      for (int nf = 0; nf < 2; ++nf)
        acc[mf][nf] = wmma16(aF[mf], bF[nf], acc[mf][nf]);

    if constexpr (!AF16) {
      if (more) commitA32(cur ^ 1, fnext);  // loadcnt wait lands after WMMAs
    }
    wait_async0();
    __syncthreads();
  }

  // Epilogue. C-frag layout: lane half h, element r -> (m = r + 8h, n = l16).
#pragma unroll
  for (int mf = 0; mf < 4; ++mf) {
#pragma unroll
    for (int nf = 0; nf < 2; ++nf) {
      const int nn = n0 + wn + nf * 16 + l16;
      const float bv = bias[nn];
#pragma unroll
      for (int r = 0; r < 8; ++r) {
        const int mm = m0 + wm + mf * 16 + (half << 3) + r;
        const float val = acc[mf][nf][r] + bv;
        if constexpr (MODE == 2) {
          out32[(size_t)mm * Dk + nn] = val;
        } else {
          const int bb = mm >> 11, ss = mm & (Sk - 1);
          const int hh = nn >> 6, hd = nn & (HDk - 1);
          const size_t idx =
              (MODE == 0) ? ((((size_t)bb * Hk + hh) * Sk + ss) * HDk + hd)
                          : ((((size_t)bb * Hk + hh) * HDk + hd) * Sk + ss);
          out16[idx] = (_Float16)val;
        }
      }
    }
  }
}

// ---------------------------------------------------------------------------
// Flash attention: block = 128 threads (4 waves) owns 64 query rows of one
// (b,h); each wave owns 16 rows. K/V 64-key blocks async-DMA'd into
// double-buffered LDS. Per wave per block: 8 WMMA scores + 8 WMMA context
// + 2 WMMA row-sums (P @ ones).
// ---------------------------------------------------------------------------
__global__ __launch_bounds__(128) void flash_attn_kernel(
    const _Float16* __restrict__ Qh, const _Float16* __restrict__ Kh,
    const _Float16* __restrict__ Vt, float* __restrict__ ctx32,
    _Float16* __restrict__ ctx16) {
  constexpr int LDT = 72;       // 144B rows, 16B aligned chunks
  constexpr int NKB = Sk / 64;  // 32 key blocks
  __shared__ _Float16 Ks[2][64 * LDT];
  __shared__ _Float16 Vs[2][64 * LDT];
  __shared__ _Float16 Ps[4 * 16 * LDT];  // per-wave P staging

  const int b = blockIdx.z, h = blockIdx.y;
  const int q0 = blockIdx.x * 64;
  const int tid = threadIdx.x;
  const int wave = tid >> 5, lid = tid & 31;
  const int half = lid >> 4, l16 = lid & 15;

  const _Float16* Qp = Qh + (((size_t)b * Hk + h) * Sk) * HDk;
  const _Float16* Kp = Kh + (((size_t)b * Hk + h) * Sk) * HDk;
  const _Float16* Vp = Vt + (((size_t)b * Hk + h) * HDk) * Sk;

  const int lr = tid >> 1;        // 0..63
  const int lc = (tid & 1) << 5;  // 0 / 32

  auto stageKV = [&](int buf, int s0) {
    const _Float16* kp = Kp + (size_t)(s0 + lr) * HDk + lc;
    const _Float16* vp = Vp + (size_t)lr * Sk + s0 + lc;
    const unsigned kd = lds_addr(&Ks[buf][lr * LDT + lc]);
    const unsigned vd = lds_addr(&Vs[buf][lr * LDT + lc]);
#pragma unroll
    for (int c = 0; c < 4; ++c) {
      async_ld_b128(kd + c * 16, kp + c * 8);
      async_ld_b128(vd + c * 16, vp + c * 8);
    }
  };

  // Q fragments persist in registers; fold softmax scale (1/8, f16-exact)
  // into Q so scores come out of WMMA pre-scaled. Packed v8h multiply keeps
  // this as clean v_pk_mul_f16 (no half-register shuffling).
  v16h qf[2];
  {
    v8h s8;
#pragma unroll
    for (int i = 0; i < 8; ++i) s8[i] = (_Float16)0.125f;
    const int qrow = q0 + wave * 16 + l16;
#pragma unroll
    for (int t = 0; t < 2; ++t) {
      H16 u;
      u.v = frag_ld(Qp + (size_t)qrow * HDk + t * 32 + (half << 3));
      u.h[0] = u.h[0] * s8;
      u.h[1] = u.h[1] * s8;
      qf[t] = u.v;
    }
  }

  v16h ones;
#pragma unroll
  for (int i = 0; i < 16; ++i) ones[i] = (_Float16)1.0f;

  float rmax[8];
  v8f o[4], lfrag;
  v8f zero = {};
#pragma unroll
  for (int r = 0; r < 8; ++r) rmax[r] = -1e30f;
#pragma unroll
  for (int nf = 0; nf < 4; ++nf) o[nf] = zero;
  lfrag = zero;

  _Float16* Pw = &Ps[wave * 16 * LDT];

  stageKV(0, 0);
  wait_async0();
  __syncthreads();

#pragma unroll 2
  for (int kb = 0; kb < NKB; ++kb) {
    const int cur = kb & 1;
    if (kb + 1 < NKB) stageKV(cur ^ 1, (kb + 1) * 64);

    const _Float16* Kb = Ks[cur];
    const _Float16* Vb = Vs[cur];

    // Preload all 8 K fragments (16 back-to-back ds_load_b128) so the
    // following 8 WMMAs get staged dscnt waits instead of serial wait-0s.
    v16h kfr[4][2];
#pragma unroll
    for (int nf = 0; nf < 4; ++nf)
#pragma unroll
      for (int t = 0; t < 2; ++t)
        kfr[nf][t] =
            frag_ld(&Kb[(nf * 16 + l16) * LDT + t * 32 + (half << 3)]);

    // S = (Q/8) K^T  (16 x 64 per wave)
    v8f sc[4];
#pragma unroll
    for (int nf = 0; nf < 4; ++nf) {
      v8f c = wmma16(qf[0], kfr[nf][0], zero);
      sc[nf] = wmma16(qf[1], kfr[nf][1], c);
    }

    // Issue V-fragment loads now: they overlap the softmax VALU work below
    // and fill the WMMA->VALU hazard slots after the score WMMAs.
    v16h vfr[4][2];
#pragma unroll
    for (int nf = 0; nf < 4; ++nf)
#pragma unroll
      for (int t = 0; t < 2; ++t)
        vfr[nf][t] =
            frag_ld(&Vb[(nf * 16 + l16) * LDT + t * 32 + (half << 3)]);

    // Row max: frag-combine then batched XOR swizzle reduce (8 rows/step).
    float pm[8];
#pragma unroll
    for (int r = 0; r < 8; ++r)
      pm[r] = fmaxf(fmaxf(sc[0][r], sc[1][r]), fmaxf(sc[2][r], sc[3][r]));
    xor_max8<1>(pm);
    xor_max8<2>(pm);
    xor_max8<4>(pm);
    xor_max8<8>(pm);

    // Online rescale + exponentiate (row sums come from WMMA below).
#pragma unroll
    for (int r = 0; r < 8; ++r) {
      const float nm = fmaxf(rmax[r], pm[r]);
      const float alpha = __expf(rmax[r] - nm);
      rmax[r] = nm;
      lfrag[r] *= alpha;
#pragma unroll
      for (int nf = 0; nf < 4; ++nf) o[nf][r] *= alpha;
#pragma unroll
      for (int nf = 0; nf < 4; ++nf) sc[nf][r] = __expf(sc[nf][r] - nm);
    }

    // Stage P as f16 in wave-private LDS (C-frag -> A-frag relayout).
    // DS ops are wave-ordered, so no barrier needed before re-reading.
#pragma unroll
    for (int nf = 0; nf < 4; ++nf)
#pragma unroll
      for (int r = 0; r < 8; ++r)
        Pw[((half << 3) + r) * LDT + nf * 16 + l16] = (_Float16)sc[nf][r];

    // O += P V and lfrag += P @ ones (row sums, no cross-lane shuffles).
    v16h pf[2];
#pragma unroll
    for (int t = 0; t < 2; ++t)
      pf[t] = frag_ld(&Pw[l16 * LDT + t * 32 + (half << 3)]);
#pragma unroll
    for (int t = 0; t < 2; ++t) {
      lfrag = wmma16(pf[t], ones, lfrag);
#pragma unroll
      for (int nf = 0; nf < 4; ++nf)
        o[nf] = wmma16(pf[t], vfr[nf][t], o[nf]);
    }

    wait_async0();
    __syncthreads();
  }

  // Finalize: ctx f32 -> d_out tail, f16 merged copy for output projection.
#pragma unroll
  for (int r = 0; r < 8; ++r) {
    const float inv = 1.0f / lfrag[r];
    const int ss = q0 + wave * 16 + (half << 3) + r;
#pragma unroll
    for (int nf = 0; nf < 4; ++nf) {
      const int hd = nf * 16 + l16;
      const float val = o[nf][r] * inv;
      ctx32[(((size_t)b * Hk + h) * Sk + ss) * HDk + hd] = val;
      ctx16[((size_t)b * Sk + ss) * Dk + h * HDk + hd] = (_Float16)val;
    }
  }
}

// ---------------------------------------------------------------------------
// Host orchestration. Workspace layout (~72 MiB):
//   [0)       4*D*D f16   converted weights (wq|wk|wv|wo)
//   [+8MiB)   B*S*D f16   Qh  [B,H,S,HD]
//   [+24MiB)  B*S*D f16   Kh  [B,H,S,HD]
//   [+40MiB)  B*S*D f16   Vt  [B,H,HD,S]
//   [+56MiB)  B*S*D f16   ctx16 merged [B,S,D]
// ---------------------------------------------------------------------------
extern "C" void kernel_launch(void* const* d_in, const int* in_sizes, int n_in,
                              void* d_out, int out_size, void* d_ws,
                              size_t ws_size, hipStream_t stream) {
  (void)in_sizes; (void)n_in; (void)out_size; (void)ws_size;
  const float* q  = (const float*)d_in[0];
  const float* k  = (const float*)d_in[1];
  const float* v  = (const float*)d_in[2];
  const float* wq = (const float*)d_in[3];
  const float* bq = (const float*)d_in[4];
  const float* wk = (const float*)d_in[5];
  const float* bk = (const float*)d_in[6];
  const float* wv = (const float*)d_in[7];
  const float* bv = (const float*)d_in[8];
  const float* wo = (const float*)d_in[9];
  const float* bo = (const float*)d_in[10];

  constexpr size_t DD = (size_t)Dk * Dk;       // 1M elems / weight
  constexpr size_t XE = (size_t)Bk * Sk * Dk;  // 8M elems / tensor
  char* ws = (char*)d_ws;
  _Float16* w16 = (_Float16*)ws;
  _Float16* Qh  = (_Float16*)(ws + 4 * DD * sizeof(_Float16));
  _Float16* Kh  = Qh + XE;
  _Float16* Vt  = Kh + XE;
  _Float16* c16 = Vt + XE;

  float* out   = (float*)d_out;
  float* ctx32 = out + XE;

  cvt_w_kernel<<<4096, 256, 0, stream>>>(wq, wk, wv, wo, w16);

  dim3 gg(Dk / 128, Mk / 128);  // (8, 64)
  gemm_xwT_kernel<0, 0><<<gg, 256, 0, stream>>>(q, w16, bq, Qh, nullptr);
  gemm_xwT_kernel<0, 0><<<gg, 256, 0, stream>>>(k, w16 + DD, bk, Kh, nullptr);
  gemm_xwT_kernel<0, 1><<<gg, 256, 0, stream>>>(v, w16 + 2 * DD, bv, Vt, nullptr);

  flash_attn_kernel<<<dim3(Sk / 64, Hk, Bk), 128, 0, stream>>>(Qh, Kh, Vt,
                                                               ctx32, c16);

  gemm_xwT_kernel<1, 2><<<gg, 256, 0, stream>>>(c16, w16 + 3 * DD, bo, nullptr,
                                                out);
}
